// SelfAttention_35244501631536
// MI455X (gfx1250) — matrix operand order repaired and checked by reading, not compiled, not run on hardware
//
#include <hip/hip_runtime.h>

typedef __attribute__((ext_vector_type(16))) __bf16 bf16x16;
typedef __attribute__((ext_vector_type(8)))  float  f32x8;
typedef __attribute__((ext_vector_type(4)))  float  f32x4;
typedef __attribute__((ext_vector_type(4)))  unsigned int u32x4;

#define BATCH 8
#define CCH   256
#define CK    32
#define NSP   4096

__device__ __forceinline__ f32x8 wmma_bf16(bf16x16 a, bf16x16 b, f32x8 c) {
  return __builtin_amdgcn_wmma_f32_16x16x32_bf16(false, a, false, b, (short)0, c, false, false);
}

// A-matrix fragment (16x32 bf16, row-major source, K contiguous per row).
// lane<16: K = k0+{0..7} and k0+{16..23}; lane>=16: +8.
__device__ __forceinline__ bf16x16 load_frag_A(const __bf16* base, int row, int stride,
                                               int k0, int lane) {
  const __bf16* p = base + (size_t)row * stride + k0 + ((lane & 16) ? 8 : 0);
  union { bf16x16 v; u32x4 q[2]; } u;
  u.q[0] = *(const u32x4*)(p);
  u.q[1] = *(const u32x4*)(p + 16);
  return u.v;
}

// B-matrix fragment (32x16 bf16) from B^T stored row-major [n][k].
// lane<16: K = k0+{0..7} and k0+{16..23}; lane>=16: +8.
__device__ __forceinline__ bf16x16 load_frag_B(const __bf16* base, int col, int stride,
                                               int k0, int lane) {
  const __bf16* p = base + (size_t)col * stride + k0 + ((lane & 16) ? 8 : 0);
  union { bf16x16 v; u32x4 q[2]; } u;
  u.q[0] = *(const u32x4*)(p);
  u.q[1] = *(const u32x4*)(p + 16);
  return u.v;
}

// A fragment built from f32 row-major source with on-the-fly bf16 convert.
__device__ __forceinline__ bf16x16 load_frag_A_f32(const float* base, int row, int stride,
                                                   int k0, int lane) {
  const float* p = base + (size_t)row * stride + k0 + ((lane & 16) ? 8 : 0);
  f32x4 a0 = *(const f32x4*)(p);
  f32x4 a1 = *(const f32x4*)(p + 4);
  f32x4 b0 = *(const f32x4*)(p + 16);
  f32x4 b1 = *(const f32x4*)(p + 20);
  bf16x16 v;
  v[0]  = (__bf16)a0[0]; v[1]  = (__bf16)a0[1]; v[2]  = (__bf16)a0[2]; v[3]  = (__bf16)a0[3];
  v[4]  = (__bf16)a1[0]; v[5]  = (__bf16)a1[1]; v[6]  = (__bf16)a1[2]; v[7]  = (__bf16)a1[3];
  v[8]  = (__bf16)b0[0]; v[9]  = (__bf16)b0[1]; v[10] = (__bf16)b0[2]; v[11] = (__bf16)b0[3];
  v[12] = (__bf16)b1[0]; v[13] = (__bf16)b1[1]; v[14] = (__bf16)b1[2]; v[15] = (__bf16)b1[3];
  return v;
}

// ---------------------------------------------------------------------------
// Pass 1: [Wf;Wg;Wh] (320x256) x x_b (256x4096) -> fT/gT (transposed bf16), hB
// ---------------------------------------------------------------------------
__global__ __launch_bounds__(256) void proj_fgh_kernel(
    const float* __restrict__ x,
    const float* __restrict__ Wf, const float* __restrict__ bf_,
    const float* __restrict__ Wg, const float* __restrict__ bg,
    const float* __restrict__ Wh, const float* __restrict__ bh,
    __bf16* __restrict__ fT, __bf16* __restrict__ gT, __bf16* __restrict__ hB) {
  const int b  = blockIdx.z;
  const int tm = blockIdx.y;            // 5 tiles of 64 rows (320 total)
  const int n0 = blockIdx.x * 128;
  const int tid = threadIdx.x, lane = tid & 31, wid = tid >> 5;
  const int wm = wid >> 2, wn = wid & 3;     // wave: 32 rows x 32 cols

  __shared__ __attribute__((aligned(16))) __bf16 xT[128 * 40];  // x^T tile [n][k]

  const int gmBase = tm * 64 + wm * 32;      // global row base for this wave
  const float* Wsrc; const float* bsrc; int ml0;
  if (gmBase < 32)      { Wsrc = Wf; bsrc = bf_; ml0 = gmBase; }
  else if (gmBase < 64) { Wsrc = Wg; bsrc = bg;  ml0 = gmBase - 32; }
  else                  { Wsrc = Wh; bsrc = bh;  ml0 = gmBase - 64; }

  const float* xb = x + (size_t)b * CCH * NSP;
  f32x8 acc[2][2] = {};

  for (int kk = 0; kk < CCH; kk += 32) {
    __syncthreads();
#pragma unroll
    for (int it = 0; it < 4; ++it) {
      int lin = it * 256 + tid;              // 1024 float4 chunks = 32k x 128n
      int k   = lin >> 5;
      int nc  = lin & 31;
      f32x4 xv = *(const f32x4*)(xb + (size_t)(kk + k) * NSP + n0 + nc * 4);
      int nl = nc * 4;
      xT[(nl + 0) * 40 + k] = (__bf16)xv[0];
      xT[(nl + 1) * 40 + k] = (__bf16)xv[1];
      xT[(nl + 2) * 40 + k] = (__bf16)xv[2];
      xT[(nl + 3) * 40 + k] = (__bf16)xv[3];
    }
    __syncthreads();
    bf16x16 afr[2], bfr[2];
#pragma unroll
    for (int mt = 0; mt < 2; ++mt)
      afr[mt] = load_frag_A_f32(Wsrc, ml0 + mt * 16 + (lane & 15), CCH, kk, lane);
#pragma unroll
    for (int nt = 0; nt < 2; ++nt)
      bfr[nt] = load_frag_B(xT, wn * 32 + nt * 16 + (lane & 15), 40, 0, lane);
#pragma unroll
    for (int mt = 0; mt < 2; ++mt)
#pragma unroll
      for (int nt = 0; nt < 2; ++nt)
        acc[mt][nt] = wmma_bf16(afr[mt], bfr[nt], acc[mt][nt]);
  }

#pragma unroll
  for (int mt = 0; mt < 2; ++mt) {
    int rbase = ml0 + mt * 16 + ((lane & 16) ? 8 : 0);
#pragma unroll
    for (int nt = 0; nt < 2; ++nt) {
      int n = n0 + wn * 32 + nt * 16 + (lane & 15);
      f32x8 v = acc[mt][nt];
      if (gmBase < 64) {
        __bf16* dst = (gmBase < 32 ? fT : gT) + ((size_t)b * NSP + n) * CK + rbase;
        union { u32x4 q; __bf16 h[8]; } u;
#pragma unroll
        for (int r = 0; r < 8; ++r) u.h[r] = (__bf16)(v[r] + bsrc[rbase + r]);
        *(u32x4*)dst = u.q;
      } else {
#pragma unroll
        for (int r = 0; r < 8; ++r) {
          int c = rbase + r;
          hB[((size_t)b * CCH + c) * NSP + n] = (__bf16)(v[r] + bsrc[c]);
        }
      }
    }
  }
}

// ---------------------------------------------------------------------------
// Pass 2a: per-row softmax stats via transposed S tiles: S^T = g^T x f
// D[j][m]: each lane owns ONE query m and 8 j values per accumulator.
// ---------------------------------------------------------------------------
__global__ __launch_bounds__(32) void attn_stats_kernel(
    const __bf16* __restrict__ fT, const __bf16* __restrict__ gT,
    float* __restrict__ Mrow, float* __restrict__ Lrow) {
  const int b  = blockIdx.y;
  const int i0 = blockIdx.x * 16;
  const int lane = threadIdx.x & 31;
  const __bf16* fTb = fT + (size_t)b * NSP * CK;
  const __bf16* gTb = gT + (size_t)b * NSP * CK;

  bf16x16 bff = load_frag_B(fTb, i0 + (lane & 15), CK, 0, lane);  // cols m (fixed)
  float mrun = -1e30f, lrun = 0.f;

  for (int J = 0; J < NSP; J += 32) {
    f32x8 z = {};
    bf16x16 ag0 = load_frag_A(gTb, J + (lane & 15), CK, 0, lane);       // rows J..J+15
    bf16x16 ag1 = load_frag_A(gTb, J + 16 + (lane & 15), CK, 0, lane);  // rows J+16..
    f32x8 s0 = wmma_bf16(ag0, bff, z);
    f32x8 s1 = wmma_bf16(ag1, bff, z);
    float bm = mrun;
#pragma unroll
    for (int r = 0; r < 8; ++r) bm = fmaxf(bm, fmaxf(s0[r], s1[r]));
    float add = 0.f;
#pragma unroll
    for (int r = 0; r < 8; ++r) add += __expf(s0[r] - bm) + __expf(s1[r] - bm);
    lrun = lrun * __expf(mrun - bm) + add;
    mrun = bm;
  }
  // lane L and lane L^16 hold the same m over disjoint j sets -> combine once
  float om = __shfl_xor(mrun, 16, 32);
  float ol = __shfl_xor(lrun, 16, 32);
  float nm = fmaxf(mrun, om);
  lrun = lrun * __expf(mrun - nm) + ol * __expf(om - nm);
  mrun = nm;
  if (lane < 16) {
    Mrow[(size_t)b * NSP + i0 + lane] = mrun;
    Lrow[(size_t)b * NSP + i0 + lane] = lrun;
  }
}

// Issue one 256x32 bf16 h tile (16KB) as async global->LDS copies.
// 4 instructions/thread, 16B per lane per instruction.
__device__ __forceinline__ void async_copy_h_tile(const __bf16* hBb, unsigned ldsBase,
                                                  int buf, int J, int tid) {
#pragma unroll
  for (int i = 0; i < 4; ++i) {
    int lin = i * 256 + tid;
    int c   = lin >> 2;            // channel row 0..255
    int ch  = lin & 3;             // 16B chunk within the 64B row
    unsigned loff = ldsBase + (unsigned)(buf * 16384 + c * 64 + ch * 16);
    unsigned goff = (unsigned)(c * (NSP * 2) + J * 2 + ch * 16);
    asm volatile("global_load_async_to_lds_b128 %0, %1, %2"
                 :: "v"(loff), "v"(goff), "s"(hBb)
                 : "memory");
  }
}

// ---------------------------------------------------------------------------
// Pass 2b: O[c,m] = sum_j h[c,j] * exp(S[m,j]-M[m])/L[m]   (flash-style)
// S computed transposed so P stores are one b128 per lane; h tiles staged in
// LDS via double-buffered async copies.
// ---------------------------------------------------------------------------
__global__ __launch_bounds__(256) void attn_apply_kernel(
    const __bf16* __restrict__ fT, const __bf16* __restrict__ gT,
    const __bf16* __restrict__ hB,
    const float* __restrict__ Mrow, const float* __restrict__ Lrow,
    __bf16* __restrict__ attHT) {
  const int b  = blockIdx.y;
  const int m0 = blockIdx.x * 32;
  const int tid = threadIdx.x, lane = tid & 31, wid = tid >> 5;

  __shared__ __attribute__((aligned(16))) __bf16 P[32 * 40];       // P[m][j]
  __shared__ __attribute__((aligned(16))) __bf16 hTile[2][256 * 32];

  const __bf16* fTb = fT + (size_t)b * NSP * CK;
  const __bf16* gTb = gT + (size_t)b * NSP * CK;
  const __bf16* hBb = hB + (size_t)b * CCH * NSP;
  const unsigned ldsBase = (unsigned)(size_t)(&hTile[0][0]);

  f32x8 acc[2][2] = {};
  const int c0 = wid * 32;                   // 8 waves cover all 256 channels

  bf16x16 bff = {};
  float Ms = 0.f, Li = 0.f;
  const int mi = wid >> 1, ji = wid & 1;     // waves 0-3 own one 16x16 S^T tile
  if (wid < 4) {
    bff = load_frag_B(fTb, m0 + mi * 16 + (lane & 15), CK, 0, lane);  // cols m
    int mrow = m0 + mi * 16 + (lane & 15);
    Ms = Mrow[(size_t)b * NSP + mrow];
    Li = 1.f / Lrow[(size_t)b * NSP + mrow];
  }

  async_copy_h_tile(hBb, ldsBase, 0, 0, tid);   // preload tile 0

  for (int J = 0; J < NSP; J += 32) {
    const int cur = (J >> 5) & 1;
    const bool more = (J + 32) < NSP;
    if (more) async_copy_h_tile(hBb, ldsBase, cur ^ 1, J + 32, tid);

    if (wid < 4) {
      f32x8 z = {};
      bf16x16 ag = load_frag_A(gTb, J + ji * 16 + (lane & 15), CK, 0, lane);
      f32x8 s = wmma_bf16(ag, bff, z);        // D[j][m]: lane m fixed, 8 j's
      union { u32x4 q; __bf16 h[8]; } u;
#pragma unroll
      for (int r = 0; r < 8; ++r) u.h[r] = (__bf16)(__expf(s[r] - Ms) * Li);
      int mloc = mi * 16 + (lane & 15);
      int jloc = ji * 16 + ((lane & 16) ? 8 : 0);
      *(u32x4*)(&P[mloc * 40 + jloc]) = u.q;  // one b128 store per lane
    }

    // wait for the tile we are about to consume (leave newest 4 in flight)
    if (more) asm volatile("s_wait_asynccnt 0x4" ::: "memory");
    else      asm volatile("s_wait_asynccnt 0x0" ::: "memory");
    __syncthreads();

    const __bf16* hL = &hTile[cur][0];
    bf16x16 ha[2], pb[2];
#pragma unroll
    for (int ct = 0; ct < 2; ++ct)
      ha[ct] = load_frag_A(hL, c0 + ct * 16 + (lane & 15), 32, 0, lane);
#pragma unroll
    for (int mh = 0; mh < 2; ++mh)
      pb[mh] = load_frag_B(P, mh * 16 + (lane & 15), 40, 0, lane);
#pragma unroll
    for (int ct = 0; ct < 2; ++ct)
#pragma unroll
      for (int mh = 0; mh < 2; ++mh)
        acc[ct][mh] = wmma_bf16(ha[ct], pb[mh], acc[ct][mh]);
    __syncthreads();
  }

  // store transposed: attHT[b][m][c] (16B-contiguous along c)
#pragma unroll
  for (int ct = 0; ct < 2; ++ct) {
    int cs = c0 + ct * 16 + ((lane & 16) ? 8 : 0);
#pragma unroll
    for (int mh = 0; mh < 2; ++mh) {
      int m = m0 + mh * 16 + (lane & 15);
      union { u32x4 q; __bf16 h[8]; } u;
#pragma unroll
      for (int r = 0; r < 8; ++r) u.h[r] = (__bf16)acc[ct][mh][r];
      *(u32x4*)(attHT + ((size_t)b * NSP + m) * CCH + cs) = u.q;
    }
  }
}

// ---------------------------------------------------------------------------
// Pass 3: out = x + gamma * (Wv x attH + bv)
// ---------------------------------------------------------------------------
__global__ __launch_bounds__(256) void out_proj_kernel(
    const float* __restrict__ x, const float* __restrict__ Wv,
    const float* __restrict__ bv, const float* __restrict__ gamma,
    const __bf16* __restrict__ attHT, float* __restrict__ out) {
  const int b  = blockIdx.z;
  const int m0 = blockIdx.y * 64;
  const int n0 = blockIdx.x * 128;
  const int tid = threadIdx.x, lane = tid & 31, wid = tid >> 5;
  const int wm = wid >> 2, wn = wid & 3;
  const float gm = gamma[0];

  const __bf16* aTb = attHT + (size_t)b * NSP * CCH;
  f32x8 acc[2][2] = {};

  for (int kk = 0; kk < CCH; kk += 32) {
    bf16x16 afr[2], bfr[2];
#pragma unroll
    for (int mt = 0; mt < 2; ++mt)
      afr[mt] = load_frag_A_f32(Wv, m0 + wm * 32 + mt * 16 + (lane & 15), CCH, kk, lane);
#pragma unroll
    for (int nt = 0; nt < 2; ++nt)
      bfr[nt] = load_frag_B(aTb, n0 + wn * 32 + nt * 16 + (lane & 15), CCH, kk, lane);
#pragma unroll
    for (int mt = 0; mt < 2; ++mt)
#pragma unroll
      for (int nt = 0; nt < 2; ++nt)
        acc[mt][nt] = wmma_bf16(afr[mt], bfr[nt], acc[mt][nt]);
  }

#pragma unroll
  for (int mt = 0; mt < 2; ++mt) {
    int ms = m0 + wm * 32 + mt * 16 + ((lane & 16) ? 8 : 0);
#pragma unroll
    for (int nt = 0; nt < 2; ++nt) {
      int n = n0 + wn * 32 + nt * 16 + (lane & 15);
#pragma unroll
      for (int r = 0; r < 8; ++r) {
        int mr = ms + r;
        size_t idx = ((size_t)b * CCH + mr) * NSP + n;
        out[idx] = x[idx] + gm * (acc[mt][nt][r] + bv[mr]);
      }
    }
  }
}

extern "C" void kernel_launch(void* const* d_in, const int* in_sizes, int n_in,
                              void* d_out, int out_size, void* d_ws, size_t ws_size,
                              hipStream_t stream) {
  (void)in_sizes; (void)n_in; (void)out_size; (void)ws_size;
  const float* x     = (const float*)d_in[0];
  const float* Wf    = (const float*)d_in[1];
  const float* bf_   = (const float*)d_in[2];
  const float* Wg    = (const float*)d_in[3];
  const float* bg    = (const float*)d_in[4];
  const float* Wh    = (const float*)d_in[5];
  const float* bh    = (const float*)d_in[6];
  const float* Wv    = (const float*)d_in[7];
  const float* bv    = (const float*)d_in[8];
  const float* gamma = (const float*)d_in[9];
  float* out = (float*)d_out;

  char* ws = (char*)d_ws;
  __bf16* fT    = (__bf16*)(ws);                                  // 2 MB
  __bf16* gT    = (__bf16*)(ws + (2ull << 20));                   // 2 MB
  __bf16* hB    = (__bf16*)(ws + (4ull << 20));                   // 16 MB
  __bf16* attHT = (__bf16*)(ws + (20ull << 20));                  // 16 MB
  float*  Mrow  = (float*)(ws + (36ull << 20));                   // 128 KB
  float*  Lrow  = (float*)(ws + (36ull << 20) + (1ull << 19));    // 128 KB

  proj_fgh_kernel<<<dim3(32, 5, BATCH), 256, 0, stream>>>(
      x, Wf, bf_, Wg, bg, Wh, bh, fT, gT, hB);
  attn_stats_kernel<<<dim3(256, BATCH), 32, 0, stream>>>(fT, gT, Mrow, Lrow);
  attn_apply_kernel<<<dim3(128, BATCH), 256, 0, stream>>>(fT, gT, hB, Mrow, Lrow, attHT);
  out_proj_kernel<<<dim3(32, 4, BATCH), 256, 0, stream>>>(x, Wv, bv, gamma, attHT, out);
}
